// PureLSTM_42150809043052
// MI455X (gfx1250) — matrix operand order repaired and checked by reading, not compiled, not run on hardware
//
#include <hip/hip_runtime.h>
#include <hip/hip_bf16.h>
#include <math.h>

typedef __attribute__((ext_vector_type(16))) __bf16 v16bf;
typedef __attribute__((ext_vector_type(8)))  __bf16 v8bf;
typedef __attribute__((ext_vector_type(8)))  float  v8f;

#define B_   32
#define T_   2048
#define IN_  256
#define H_   512
#define G4_  2048      // 4*H
#define OUT_ 256
#define L_   2
#define MROWS (B_*T_)  // 65536

// D = A(16x32 bf16) * B(32x16 bf16) + C(16x16 f32)
__device__ __forceinline__ v8f wmma_bf16(v16bf a, v16bf b, v8f c) {
  return __builtin_amdgcn_wmma_f32_16x16x32_bf16(false, a, false, b, (short)0, c, false, false);
}

__device__ __forceinline__ v16bf cat8(v8bf lo, v8bf hi) {
  return __builtin_shufflevector(lo, hi, 0,1,2,3,4,5,6,7,8,9,10,11,12,13,14,15);
}

__device__ __forceinline__ float sigmoidf_(float x) {
  return 1.0f / (1.0f + __expf(-x));
}

// async global -> LDS copy of 16 bytes per lane (CDNA5, ASYNCcnt-tracked)
__device__ __forceinline__ void async_g2l_b128(unsigned lds_off, const void* gaddr) {
  asm volatile("global_load_async_to_lds_b128 %0, %1, off"
               :: "v"(lds_off), "v"((unsigned long long)(uintptr_t)gaddr)
               : "memory");
}
__device__ __forceinline__ void wait_asynccnt0() {
  asm volatile("s_wait_asynccnt 0x0" ::: "memory");
}

// ---------------- elementwise f32 -> bf16 ----------------
__global__ void cvt_bf16_kernel(const float* __restrict__ src, __bf16* __restrict__ dst, int n) {
  int i = blockIdx.x * 256 + threadIdx.x;
  if (i < n) dst[i] = (__bf16)src[i];
}

// ---------------- pack row-major [K,N] f32 weight into WMMA B-fragment blocks ----
// Block (ntile,ktile): 512 bf16, per-lane 16 contiguous halves.
// Element (k,n): lane = (n&15) + 16*((k>>3)&1), elem = (k&7) + 8*((k>>4)&1)  (ISA 7.12.2 B layout)
__global__ void pack_b_kernel(const float* __restrict__ W, __bf16* __restrict__ out, int K, int N) {
  int i = blockIdx.x * 256 + threadIdx.x;
  if (i >= K * N) return;
  int k = i / N, n = i % N;
  int lane = (n & 15) + (((k >> 3) & 1) << 4);
  int e    = (k & 7) + (((k >> 4) & 1) << 3);
  size_t blk = (size_t)(n >> 4) * (size_t)(K >> 5) + (size_t)(k >> 5);
  out[blk * 512 + (size_t)lane * 16 + e] = (__bf16)W[i];
}

// ---------------- WMMA GEMM: out[M,N] = A[M,K](bf16,rowmajor) * Bpacked + bias ----
// 128 threads = 4 waves, each wave owns a 16x64 output tile.
template<bool OUT_BF16>
__global__ void gemm_bias_kernel(const __bf16* __restrict__ A, const __bf16* __restrict__ Bp,
                                 const float* __restrict__ bias, void* __restrict__ out,
                                 int M, int N, int K) {
  const int lane = threadIdx.x & 31;
  const int wave = threadIdx.x >> 5;
  const int m0 = blockIdx.x << 4;
  const int n0 = (blockIdx.y << 8) + (wave << 6);
  const int kTiles = K >> 5;
  (void)M;

  v8f zero = {};
  v8f acc[4];
  #pragma unroll
  for (int j = 0; j < 4; ++j) acc[j] = zero;

  const __bf16* arow = A + (size_t)(m0 + (lane & 15)) * K + ((lane >> 4) << 3);
  for (int kt = 0; kt < kTiles; ++kt) {
    const __bf16* ap = arow + (kt << 5);
    v8bf lo = *(const v8bf*)(ap);
    v8bf hi = *(const v8bf*)(ap + 16);
    v16bf a = cat8(lo, hi);
    #pragma unroll
    for (int j = 0; j < 4; ++j) {
      size_t ntile = (size_t)((n0 >> 4) + j);
      const v16bf b = *(const v16bf*)(Bp + (ntile * (size_t)kTiles + kt) * 512 + (size_t)lane * 16);
      acc[j] = wmma_bf16(a, b, acc[j]);
    }
  }
  // D layout: lane holds n = n0+lane%16, rows m = r + 8*(lane>>4), r = VGPR index
  #pragma unroll
  for (int j = 0; j < 4; ++j) {
    int n = n0 + (j << 4) + (lane & 15);
    float bv = bias[n];
    #pragma unroll
    for (int r = 0; r < 8; ++r) {
      int m = m0 + r + ((lane >> 4) << 3);
      float v = acc[j][r] + bv;
      if (OUT_BF16) ((__bf16*)out)[(size_t)m * N + n] = (__bf16)v;
      else          ((float*)out)[(size_t)m * N + n]  = v;
    }
  }
}

// ---------------- persistent LSTM recurrence ----------------
// grid = 2 (batch tiles of 16 rows), block = 512 (16 waves), ONE block per WGP
// (304KB LDS) -> __launch_bounds__(512,1) frees the full VGPR budget (4 waves/SIMD).
// Wave w owns gate columns [w*128, w*128+128).
// xp : [B,T,4H] f32 precomputed input contribution (bias folded in)
// Whp: packed recurrent weights for this layer (K=512, N=2048)
// hseq: [B,T,H] bf16, written per timestep (layer output / next layer input)
// LDS: h(16KB) | c(32KB) | gates(128KB) | xp_stage(128KB) = 304KB (CDNA5 WGP only)
__global__ void __launch_bounds__(512, 1)
lstm_scan_kernel(const float* __restrict__ xp, const __bf16* __restrict__ Whp,
                 __bf16* __restrict__ hseq) {
  extern __shared__ char smem[];
  __bf16* h_cur    = (__bf16*)smem;                                     // 16 x 512 bf16
  float*  c_st     = (float*)(smem + 16*H_*2);                          // 16 x 512 f32
  float*  gates    = (float*)(smem + 16*H_*2 + 16*H_*4);                // 16 x 2048 f32
  float*  xp_stage = (float*)(smem + 16*H_*2 + 16*H_*4 + 16*G4_*4);     // 16 x 2048 f32

  const int tid  = threadIdx.x;
  const int lane = tid & 31;
  const int wave = tid >> 5;          // 0..15
  const int m0   = blockIdx.x << 4;   // batch tile base
  const int n0w  = wave << 7;         // 128 gate columns per wave

  for (int i = tid; i < 16 * H_; i += 512) { h_cur[i] = (__bf16)0.f; c_st[i] = 0.f; }
  const unsigned lds_stage = (unsigned)(uintptr_t)(void*)xp_stage;
  __syncthreads();

  const int aRow = lane & 15;
  const int aOff = (lane >> 4) << 3;
  // per-lane base of this wave's B fragments: frag(j, kt) at bbase + (j*16 + kt)*512
  const __bf16* bbase = Whp + (size_t)(n0w >> 4) * 16 * 512 + (size_t)lane * 16;

  for (int t = 0; t < T_; ++t) {
    // ---- kick off async prefetch of xp[:, t, :] into LDS (overlaps the WMMAs) ----
    #pragma unroll
    for (int m = 0; m < 16; ++m) {
      const float* g = xp + ((size_t)(m0 + m) * T_ + t) * (size_t)G4_ + (tid << 2);
      unsigned l = lds_stage + (unsigned)(((m << 11) + (tid << 2)) << 2);
      async_g2l_b128(l, g);
    }

    v8f zero = {};
    v8f acc[8];
    #pragma unroll
    for (int j = 0; j < 8; ++j) acc[j] = zero;

    // gates[16, n0w:n0w+128] = h_cur[16,512] @ Wh[512, n0w:n0w+128]
    // Software-pipelined: B fragments double-buffered in groups of 4 n-tiles,
    // so loads for the next group are in flight during the current 4 WMMAs.
    v16bf bA[4], bB[4];
    #pragma unroll
    for (int j = 0; j < 4; ++j) bA[j] = *(const v16bf*)(bbase + (size_t)j * 16 * 512);
    v16bf a;
    {
      const __bf16* hp = h_cur + aRow * H_ + aOff;
      a = cat8(*(const v8bf*)hp, *(const v8bf*)(hp + 16));
    }

    #pragma unroll
    for (int kt = 0; kt < (H_ >> 5); ++kt) {
      #pragma unroll
      for (int j = 0; j < 4; ++j)
        bB[j] = *(const v16bf*)(bbase + ((size_t)(4 + j) * 16 + kt) * 512);
      #pragma unroll
      for (int j = 0; j < 4; ++j) acc[j] = wmma_bf16(a, bA[j], acc[j]);

      v16bf an = a;
      if (kt < (H_ >> 5) - 1) {
        #pragma unroll
        for (int j = 0; j < 4; ++j)
          bA[j] = *(const v16bf*)(bbase + ((size_t)j * 16 + kt + 1) * 512);
        const __bf16* hp = h_cur + aRow * H_ + ((kt + 1) << 5) + aOff;
        an = cat8(*(const v8bf*)hp, *(const v8bf*)(hp + 16));
      }
      #pragma unroll
      for (int j = 0; j < 4; ++j) acc[4 + j] = wmma_bf16(a, bB[j], acc[4 + j]);
      a = an;
    }

    #pragma unroll
    for (int j = 0; j < 8; ++j) {
      int n = n0w + (j << 4) + (lane & 15);
      #pragma unroll
      for (int r = 0; r < 8; ++r) {
        int m = r + ((lane >> 4) << 3);
        gates[m * G4_ + n] = acc[j][r];
      }
    }
    wait_asynccnt0();      // xp stage resident in LDS
    __syncthreads();

    // cell update: 16 rows x 512 hidden, processed as adjacent pairs (b64 LDS ops,
    // packed b32 stores). 8 pairs per thread.
    #pragma unroll
    for (int p = 0; p < 8; ++p) {
      int q  = p * 512 + tid;          // pair id 0..4095
      int m  = q >> 8;                 // 256 pairs per row
      int jp = (q & 255) << 1;
      int g0 = m * G4_ + jp;
      float2 xi = *(const float2*)&xp_stage[g0];
      float2 xf = *(const float2*)&xp_stage[g0 + H_];
      float2 xg = *(const float2*)&xp_stage[g0 + 2 * H_];
      float2 xo = *(const float2*)&xp_stage[g0 + 3 * H_];
      float2 gi = *(const float2*)&gates[g0];
      float2 gf = *(const float2*)&gates[g0 + H_];
      float2 gg = *(const float2*)&gates[g0 + 2 * H_];
      float2 go = *(const float2*)&gates[g0 + 3 * H_];
      float2 co = *(const float2*)&c_st[m * H_ + jp];
      float c0 = sigmoidf_(gf.x + xf.x) * co.x + sigmoidf_(gi.x + xi.x) * tanhf(gg.x + xg.x);
      float c1 = sigmoidf_(gf.y + xf.y) * co.y + sigmoidf_(gi.y + xi.y) * tanhf(gg.y + xg.y);
      float h0 = sigmoidf_(go.x + xo.x) * tanhf(c0);
      float h1 = sigmoidf_(go.y + xo.y) * tanhf(c1);
      *(float2*)&c_st[m * H_ + jp] = make_float2(c0, c1);
      union { __bf16 b[2]; unsigned u; } hu;
      hu.b[0] = (__bf16)h0; hu.b[1] = (__bf16)h1;
      *(unsigned*)&h_cur[m * H_ + jp] = hu.u;
      *(unsigned*)((__bf16*)hseq + ((size_t)(m0 + m) * T_ + t) * (size_t)H_ + jp) = hu.u;
    }
    __syncthreads();
  }
}

// ---------------- host orchestration ----------------
extern "C" void kernel_launch(void* const* d_in, const int* in_sizes, int n_in,
                              void* d_out, int out_size, void* d_ws, size_t ws_size,
                              hipStream_t stream) {
  (void)in_sizes; (void)n_in; (void)out_size; (void)ws_size;
  const float* x     = (const float*)d_in[0];
  const float* W_in  = (const float*)d_in[1];
  const float* b_in  = (const float*)d_in[2];
  const float* Wx    = (const float*)d_in[3];
  const float* Wh    = (const float*)d_in[4];
  const float* bgate = (const float*)d_in[5];
  const float* W_out = (const float*)d_in[6];
  const float* b_out = (const float*)d_in[7];

  char* ws = (char*)d_ws;
  size_t off = 0;
  auto alloc = [&](size_t bytes) -> void* {
    void* p = ws + off;
    off += (bytes + 255) & ~(size_t)255;
    return p;
  };
  float*  xp     = (float*) alloc((size_t)MROWS * G4_ * 4);   // 512 MB gate preacts
  __bf16* x_bf   = (__bf16*)alloc((size_t)MROWS * IN_ * 2);
  __bf16* h_seq  = (__bf16*)alloc((size_t)MROWS * H_  * 2);
  __bf16* Win_p  = (__bf16*)alloc((size_t)IN_ * H_  * 2);
  __bf16* Wx_p   = (__bf16*)alloc((size_t)L_ * H_ * G4_ * 2);
  __bf16* Wh_p   = (__bf16*)alloc((size_t)L_ * H_ * G4_ * 2);
  __bf16* Wout_p = (__bf16*)alloc((size_t)H_ * OUT_ * 2);

  // 1) convert x to bf16; pack all weights into WMMA B layout
  cvt_bf16_kernel<<<(MROWS * IN_ + 255) / 256, 256, 0, stream>>>(x, x_bf, MROWS * IN_);
  pack_b_kernel<<<(IN_ * H_ + 255) / 256, 256, 0, stream>>>(W_in, Win_p, IN_, H_);
  for (int l = 0; l < L_; ++l) {
    pack_b_kernel<<<(H_ * G4_ + 255) / 256, 256, 0, stream>>>(Wx + (size_t)l * H_ * G4_,
                                                              Wx_p + (size_t)l * H_ * G4_, H_, G4_);
    pack_b_kernel<<<(H_ * G4_ + 255) / 256, 256, 0, stream>>>(Wh + (size_t)l * H_ * G4_,
                                                              Wh_p + (size_t)l * H_ * G4_, H_, G4_);
  }
  pack_b_kernel<<<(H_ * OUT_ + 255) / 256, 256, 0, stream>>>(W_out, Wout_p, H_, OUT_);

  // 2) input linear: h_seq = x @ W_in + b_in (bf16 out)
  gemm_bias_kernel<true><<<dim3(MROWS / 16, H_ / 256), 128, 0, stream>>>(
      x_bf, Win_p, b_in, h_seq, MROWS, H_, IN_);

  // 3) per-layer: big gate GEMM then persistent recurrence
  const int scan_smem = 16 * H_ * 2 + 16 * H_ * 4 + 16 * G4_ * 4 + 16 * G4_ * 4;  // 304 KB
  (void)hipFuncSetAttribute(reinterpret_cast<const void*>(lstm_scan_kernel),
                            hipFuncAttributeMaxDynamicSharedMemorySize, scan_smem);
  for (int l = 0; l < L_; ++l) {
    gemm_bias_kernel<false><<<dim3(MROWS / 16, G4_ / 256), 128, 0, stream>>>(
        h_seq, Wx_p + (size_t)l * H_ * G4_, bgate + (size_t)l * G4_, xp, MROWS, G4_, H_);
    lstm_scan_kernel<<<B_ / 16, 512, scan_smem, stream>>>(
        xp, Wh_p + (size_t)l * H_ * G4_, h_seq);
  }

  // 4) output linear: d_out = h_seq @ W_out + b_out (f32 out)
  gemm_bias_kernel<false><<<dim3(MROWS / 16, OUT_ / 256), 128, 0, stream>>>(
      h_seq, Wout_p, b_out, d_out, MROWS, OUT_, H_);
}
